// TemporalRUSMoELayer_23648089932609
// MI455X (gfx1250) — matrix operand order repaired and checked by reading, not compiled, not run on hardware
//
#include <hip/hip_runtime.h>
#include <hip/hip_bf16.h>
#include <math.h>

// ---------------------------------------------------------------------------
// Types for CDNA5 WMMA
// ---------------------------------------------------------------------------
typedef __attribute__((ext_vector_type(16))) __bf16 v16bf;
typedef __attribute__((ext_vector_type(2)))  __bf16 v2bf;
typedef __attribute__((ext_vector_type(8)))  float  v8f;

#define GF_RELU 1
#define GF_ACC  2

__device__ __forceinline__ __bf16 bits2bf(unsigned short s) {
  __bf16 r; __builtin_memcpy(&r, &s, sizeof(r)); return r;
}
// float -> bf16 via native clang conversion (backend selects HW cvt if present)
__device__ __forceinline__ unsigned short f2bfbits(float f) {
  __bf16 b = (__bf16)f;
  unsigned short s; __builtin_memcpy(&s, &b, sizeof(s));
  return s;
}
__device__ __forceinline__ __bf16 f2bf(float f) { return (__bf16)f; }

// pack two floats into one dword of bf16 (v_cvt_pk_bf16_f32 when available)
__device__ __forceinline__ unsigned pk2(float a, float b) {
#if __has_builtin(__builtin_amdgcn_cvt_pk_bf16_f32)
  v2bf p = __builtin_amdgcn_cvt_pk_bf16_f32(a, b);
#else
  v2bf p; p[0] = (__bf16)a; p[1] = (__bf16)b;
#endif
  unsigned r; __builtin_memcpy(&r, &p, sizeof(r));
  return r;
}
// unpack a dword (two bf16) into fragment elements 2v, 2v+1
__device__ __forceinline__ void put_pair(v16bf& dst, int v, unsigned pr) {
  dst[2*v]   = bits2bf((unsigned short)(pr & 0xFFFFu));
  dst[2*v+1] = bits2bf((unsigned short)(pr >> 16));
}
__device__ __forceinline__ void put_pairf(v16bf& dst, int v, float a, float b) {
  put_pair(dst, v, pk2(a, b));
}

__device__ __forceinline__ float redsum32(float v) {
#pragma unroll
  for (int d = 1; d < 32; d <<= 1) v += __shfl_xor(v, d, 32);
  return v;
}
__device__ __forceinline__ float redsum16(float v) {
#pragma unroll
  for (int d = 1; d < 16; d <<= 1) v += __shfl_xor(v, d, 32);
  return v;
}
__device__ __forceinline__ float redmax16(float v) {
#pragma unroll
  for (int d = 1; d < 16; d <<= 1) v = fmaxf(v, __shfl_xor(v, d, 32));
  return v;
}

// ---------------------------------------------------------------------------
// WMMA GEMM:  C[N,O] = act(A[N,K] * W[O,K]^T + bias) (opt rowscale / accum)
// Block tile 64x128, 128 threads = 4 waves, each wave 32x64 (2x4 WMMA tiles).
// All launch shapes are exact multiples (N%64==0, O%128==0, K%32==0) so the
// staging path is unguarded float4 loads -> packed bf16 uint2 LDS stores.
// ---------------------------------------------------------------------------
__global__ __launch_bounds__(128) void gemm_wmma_kernel(
    const float* __restrict__ A, int lda,
    const float* __restrict__ W, int ldw,
    const float* __restrict__ bias,
    const float* __restrict__ rowscale, int rs_stride,
    float* __restrict__ C, int ldc,
    int K, int flags)
{
  __shared__ unsigned short As[64][36];
  __shared__ unsigned short Bs[128][36];
  const int tid  = threadIdx.x;
  const int wave = tid >> 5;
  const int lane = tid & 31;
  const int half = lane >> 4;
  const int l15  = lane & 15;
  const int bm   = blockIdx.x * 64;    // N offset
  const int bo   = blockIdx.y * 128;   // O offset
  const int wmo  = (wave & 1) * 32;    // wave m offset in tile
  const int wno  = (wave >> 1) * 64;   // wave n offset in tile

  v8f cc[2][4];
#pragma unroll
  for (int i = 0; i < 2; ++i)
#pragma unroll
    for (int j = 0; j < 4; ++j)
#pragma unroll
      for (int r = 0; r < 8; ++r) cc[i][j][r] = 0.f;

  for (int k0 = 0; k0 < K; k0 += 32) {
    // stage A (64x32) and W (128x32) as bf16, vectorized
#pragma unroll
    for (int i = tid; i < 512; i += 128) {
      int r = i >> 3, c4 = (i & 7) * 4;
      float4 va = *(const float4*)(A + (size_t)(bm + r) * lda + k0 + c4);
      *(uint2*)&As[r][c4] = make_uint2(pk2(va.x, va.y), pk2(va.z, va.w));
    }
#pragma unroll
    for (int i = tid; i < 1024; i += 128) {
      int r = i >> 3, c4 = (i & 7) * 4;
      float4 vb = *(const float4*)(W + (size_t)(bo + r) * ldw + k0 + c4);
      *(uint2*)&Bs[r][c4] = make_uint2(pk2(vb.x, vb.y), pk2(vb.z, vb.w));
    }
    __syncthreads();

    v16bf a0, a1, bb[4];
#pragma unroll
    for (int v = 0; v < 8; ++v) {
      // A layout (16x32, 16-bit): K = (v/4)*16 + half*8 + (v%4)*2 + pos
      int ak = ((v >> 2) << 4) + (half << 3) + ((v & 3) << 1);
      put_pair(a0, v, *(const unsigned*)&As[wmo + l15][ak]);
      put_pair(a1, v, *(const unsigned*)&As[wmo + 16 + l15][ak]);
      // B layout (32x16, 16-bit): K = (v/4)*16 + half*8 + (v%4)*2 + pos ; N = l15
      int bk = ((v >> 2) << 4) + (half << 3) + ((v & 3) << 1);
#pragma unroll
      for (int j = 0; j < 4; ++j)
        put_pair(bb[j], v, *(const unsigned*)&Bs[wno + j * 16 + l15][bk]);
    }
#pragma unroll
    for (int j = 0; j < 4; ++j) {
      cc[0][j] = __builtin_amdgcn_wmma_f32_16x16x32_bf16(false, a0, false, bb[j], (short)0, cc[0][j], false, false);
      cc[1][j] = __builtin_amdgcn_wmma_f32_16x16x32_bf16(false, a1, false, bb[j], (short)0, cc[1][j], false, false);
    }
    __syncthreads();
  }

  // epilogue: C layout — M = r + half*8, N = l15
#pragma unroll
  for (int ti = 0; ti < 2; ++ti) {
#pragma unroll
    for (int tj = 0; tj < 4; ++tj) {
      int col = bo + wno + tj * 16 + l15;
      float bv = bias ? bias[col] : 0.f;
#pragma unroll
      for (int r = 0; r < 8; ++r) {
        int row = bm + wmo + ti * 16 + r + half * 8;
        float v = cc[ti][tj][r] + bv;
        if (flags & GF_RELU) v = fmaxf(v, 0.f);
        if (rowscale) v *= rowscale[(size_t)row * rs_stride];
        if (flags & GF_ACC) C[(size_t)row * ldc + col] += v;
        else                C[(size_t)row * ldc + col] = v;
      }
    }
  }
}

// ---------------------------------------------------------------------------
// Flash attention for a synergy expert. qkv: [N,768] (q|k|v each 256 = 4x64).
// Block: 128 threads = 4 waves = 64 query rows, one head per blockIdx.y.
// K/V tiles (32 keys) staged cooperatively in LDS as bf16 (V transposed so
// P*V fragment pairs are contiguous). Key-masked online softmax; WMMA for
// both Q*K^T and P*V.
// ---------------------------------------------------------------------------
__global__ __launch_bounds__(128) void flash_attn_kernel(
    const float* __restrict__ qkv,
    const float* __restrict__ routed,   // [N] 0/1
    float* __restrict__ ctx2,           // [N,256]
    int N)
{
  const int h    = blockIdx.y;
  const int tid  = threadIdx.x;
  const int wave = tid >> 5;
  const int lane = tid & 31;
  const int half = lane >> 4;
  const int l15  = lane & 15;
  const int q0   = blockIdx.x * 64 + wave * 16;

  __shared__ unsigned short Kt[32][68];     // [key][dh]  bf16
  __shared__ unsigned short Vt[64][36];     // [dh][key]  bf16 (transposed)
  __shared__ unsigned short Pw[4][16][36];  // per-wave P tile, A-layout source

  // Q fragments (16x64 -> two 16x32 A fragments), held in registers
  v16bf aq0, aq1;
  const float* qp = qkv + (size_t)(q0 + l15) * 768 + h * 64;
#pragma unroll
  for (int v = 0; v < 8; ++v) {
    int ak = ((v >> 2) << 4) + (half << 3) + ((v & 3) << 1);
    float2 qa = *(const float2*)(qp + ak);
    float2 qb = *(const float2*)(qp + 32 + ak);
    put_pairf(aq0, v, qa.x, qa.y);
    put_pairf(aq1, v, qb.x, qb.y);
  }

  float mrow[8], lrow[8];
  v8f oo[4];
#pragma unroll
  for (int r = 0; r < 8; ++r) {
    mrow[r] = -3.0e38f; lrow[r] = 0.f;
    oo[0][r] = 0.f; oo[1][r] = 0.f; oo[2][r] = 0.f; oo[3][r] = 0.f;
  }

  for (int j0 = 0; j0 < N; j0 += 32) {
    // --- cooperative K/V tile staging (32 keys x 64 dh, float4 loads) ---
#pragma unroll
    for (int i = tid; i < 512; i += 128) {
      int key = i >> 4, c4 = (i & 15) * 4;
      const float* kp = qkv + (size_t)(j0 + key) * 768 + 256 + h * 64 + c4;
      float4 kv = *(const float4*)kp;
      *(uint2*)&Kt[key][c4] = make_uint2(pk2(kv.x, kv.y), pk2(kv.z, kv.w));
      float4 vv = *(const float4*)(kp + 256);
      unsigned vp01 = pk2(vv.x, vv.y);
      unsigned vp23 = pk2(vv.z, vv.w);
      Vt[c4 + 0][key] = (unsigned short)(vp01 & 0xFFFFu);
      Vt[c4 + 1][key] = (unsigned short)(vp01 >> 16);
      Vt[c4 + 2][key] = (unsigned short)(vp23 & 0xFFFFu);
      Vt[c4 + 3][key] = (unsigned short)(vp23 >> 16);
    }
    __syncthreads();

    // --- S = Q * K^T for 32 keys: two 16x16 C fragments ---
    v8f s0, s1;
#pragma unroll
    for (int r = 0; r < 8; ++r) { s0[r] = 0.f; s1[r] = 0.f; }
#pragma unroll
    for (int dc = 0; dc < 2; ++dc) {
      v16bf bk0, bk1;
#pragma unroll
      for (int v = 0; v < 8; ++v) {
        int kk = dc * 32 + ((v >> 2) << 4) + (half << 3) + ((v & 3) << 1);   // dh index
        put_pair(bk0, v, *(const unsigned*)&Kt[l15][kk]);
        put_pair(bk1, v, *(const unsigned*)&Kt[16 + l15][kk]);
      }
      if (dc == 0) {
        s0 = __builtin_amdgcn_wmma_f32_16x16x32_bf16(false, aq0, false, bk0, (short)0, s0, false, false);
        s1 = __builtin_amdgcn_wmma_f32_16x16x32_bf16(false, aq0, false, bk1, (short)0, s1, false, false);
      } else {
        s0 = __builtin_amdgcn_wmma_f32_16x16x32_bf16(false, aq1, false, bk0, (short)0, s0, false, false);
        s1 = __builtin_amdgcn_wmma_f32_16x16x32_bf16(false, aq1, false, bk1, (short)0, s1, false, false);
      }
    }

    // --- scale + key mask + online softmax (rows live in 16-lane groups) ---
    float msk0 = (routed[j0 + l15]      != 0.f) ? 0.f : -1e9f;
    float msk1 = (routed[j0 + 16 + l15] != 0.f) ? 0.f : -1e9f;
#pragma unroll
    for (int r = 0; r < 8; ++r) {
      s0[r] = s0[r] * 0.125f + msk0;
      s1[r] = s1[r] * 0.125f + msk1;
      float tm = redmax16(fmaxf(s0[r], s1[r]));
      float mnew = fmaxf(mrow[r], tm);
      float corr = __expf(mrow[r] - mnew);
      mrow[r] = mnew;
      lrow[r] *= corr;
      oo[0][r] *= corr; oo[1][r] *= corr; oo[2][r] *= corr; oo[3][r] *= corr;
      float p0v = __expf(s0[r] - mnew);
      float p1v = __expf(s1[r] - mnew);
      s0[r] = p0v; s1[r] = p1v;
      lrow[r] += redsum16(p0v + p1v);
    }

    // --- transpose P (C layout) into A-fragment layout via per-wave LDS ---
#pragma unroll
    for (int r = 0; r < 8; ++r) {
      Pw[wave][r + half * 8][l15]      = f2bfbits(s0[r]);
      Pw[wave][r + half * 8][16 + l15] = f2bfbits(s1[r]);
    }
    v16bf pa;   // same-wave DS ordering makes this RAW safe without a barrier
#pragma unroll
    for (int v = 0; v < 8; ++v) {
      int ak = ((v >> 2) << 4) + (half << 3) + ((v & 3) << 1);
      put_pair(pa, v, *(const unsigned*)&Pw[wave][l15][ak]);
    }

    // --- O += P * V : four dh sub-tiles of 16 (pairs contiguous in Vt) ---
#pragma unroll
    for (int nt = 0; nt < 4; ++nt) {
      v16bf bv;
#pragma unroll
      for (int v = 0; v < 8; ++v) {
        int kk = ((v >> 2) << 4) + (half << 3) + ((v & 3) << 1);             // key index
        put_pair(bv, v, *(const unsigned*)&Vt[nt * 16 + l15][kk]);
      }
      oo[nt] = __builtin_amdgcn_wmma_f32_16x16x32_bf16(false, pa, false, bv, (short)0, oo[nt], false, false);
    }
    __syncthreads();
  }

  // --- normalize and store ctx ---
#pragma unroll
  for (int r = 0; r < 8; ++r) {
    int row = q0 + r + half * 8;
    float inv = 1.0f / lrow[r];
    float* op = ctx2 + (size_t)row * 256 + h * 64 + l15;
    op[0]  = oo[0][r] * inv;
    op[16] = oo[1][r] * inv;
    op[32] = oo[2][r] * inv;
    op[48] = oo[3][r] * inv;
  }
}

// ---------------------------------------------------------------------------
// Router neighbor-attention: one wave per token (lane = DK = DV = 32 dim)
// ---------------------------------------------------------------------------
__global__ __launch_bounds__(256) void router_attn_kernel(
    const float* __restrict__ comb,  // proc in cols [0,128)
    const float* __restrict__ R, const float* __restrict__ S,
    const float* __restrict__ qw, const float* __restrict__ qb,
    const float* __restrict__ kw, const float* __restrict__ kb,
    const float* __restrict__ vw, const float* __restrict__ vb,
    float* __restrict__ ctxout)      // [N,32]
{
  int token = blockIdx.x * 8 + (threadIdx.x >> 5);
  int lane  = threadIdx.x & 31;
  int b = token >> 10;
  int m = (token >> 8) & 3;
  int t = token & 255;

  float q = qb[lane];
  const float* pr = comb + (size_t)token * 256;
  for (int i = 0; i < 128; ++i) q += pr[i] * qw[lane * 128 + i];

  float sc[3], Rv[3], Sv[3];
  int jj = 0;
  for (int j = 0; j < 4; ++j) {
    if (j == m) continue;
    size_t idx = ((size_t)((b * 4 + m) * 4 + j)) * 256 + t;
    Rv[jj] = R[idx]; Sv[jj] = S[idx];
    float kd = kw[lane * 2] * Rv[jj] + kw[lane * 2 + 1] * Sv[jj] + kb[lane];
    sc[jj] = redsum32(q * kd) * 0.1767766953f;   // 1/sqrt(32)
    ++jj;
  }
  float mx = fmaxf(sc[0], fmaxf(sc[1], sc[2]));
  float e0 = __expf(sc[0] - mx), e1 = __expf(sc[1] - mx), e2 = __expf(sc[2] - mx);
  float inv = 1.f / (e0 + e1 + e2);
  float aw0 = e0 * inv, aw1 = e1 * inv, aw2 = e2 * inv;
  float c = 0.f;
  c += aw0 * (vw[lane * 2] * Rv[0] + vw[lane * 2 + 1] * Sv[0] + vb[lane]);
  c += aw1 * (vw[lane * 2] * Rv[1] + vw[lane * 2 + 1] * Sv[1] + vb[lane]);
  c += aw2 * (vw[lane * 2] * Rv[2] + vw[lane * 2 + 1] * Sv[2] + vb[lane]);
  ctxout[(size_t)token * 32 + lane] = c;
}

// ---------------------------------------------------------------------------
// GRU scan: one block per sequence (B*M = 8), 384 threads (= 3*GH gates)
// ---------------------------------------------------------------------------
__global__ __launch_bounds__(384) void gru_kernel(
    const float* __restrict__ U,    // [8,256]
    const float* __restrict__ ctx,  // [N,32]
    const float* __restrict__ wih,  // [384,33]
    const float* __restrict__ whh,  // [384,128]
    const float* __restrict__ bih, const float* __restrict__ bhh,
    float* __restrict__ comb)       // writes cols [128,256)
{
  int bm = blockIdx.x;
  int g  = threadIdx.x;
  __shared__ float h[128];
  __shared__ float xt[33];
  __shared__ float giS[384], ghS[384];
  if (g < 128) h[g] = 0.f;
  __syncthreads();
  for (int t = 0; t < 256; ++t) {
    int token = bm * 256 + t;
    if (g == 0) xt[0] = U[token];
    if (g < 32) xt[1 + g] = ctx[(size_t)token * 32 + g];
    __syncthreads();
    float gi = bih[g], gh = bhh[g];
    for (int i = 0; i < 33; ++i)  gi += wih[g * 33 + i] * xt[i];
    for (int i = 0; i < 128; ++i) gh += whh[g * 128 + i] * h[i];
    giS[g] = gi; ghS[g] = gh;
    __syncthreads();
    if (g < 128) {
      float r = 1.f / (1.f + __expf(-(giS[g] + ghS[g])));
      float z = 1.f / (1.f + __expf(-(giS[128 + g] + ghS[128 + g])));
      float n = tanhf(giS[256 + g] + r * ghS[256 + g]);
      float hn = (1.f - z) * n + z * h[g];
      h[g] = hn;
      comb[(size_t)token * 256 + 128 + g] = hn;
    }
    __syncthreads();
  }
}

// ---------------------------------------------------------------------------
// Router MLP second layer (O=8, tiny): one thread per (token, expert)
// ---------------------------------------------------------------------------
__global__ void logits_kernel(const float* __restrict__ hidden,  // [N,128]
                              const float* __restrict__ w2,      // [8,128]
                              const float* __restrict__ b2,
                              float* __restrict__ logits)        // [N,8]
{
  int idx = blockIdx.x * blockDim.x + threadIdx.x;
  if (idx >= 2048 * 8) return;
  int tkn = idx >> 3, e = idx & 7;
  float acc = b2[e];
  const float* hp = hidden + (size_t)tkn * 128;
  for (int i = 0; i < 128; ++i) acc += hp[i] * w2[e * 128 + i];
  logits[idx] = acc;
}

// ---------------------------------------------------------------------------
// Softmax gating + top-2 routing; emits gating, topi (as float), w_all, masks
// ---------------------------------------------------------------------------
__global__ void gate_topk_kernel(const float* __restrict__ logits,
                                 float* __restrict__ gate_o,
                                 float* __restrict__ topi_o,
                                 float* __restrict__ w_all,
                                 float* __restrict__ routed)
{
  int n = blockIdx.x * blockDim.x + threadIdx.x;
  if (n >= 2048) return;
  float lg[8];
  float mx = -3e38f;
  for (int e = 0; e < 8; ++e) { lg[e] = logits[n * 8 + e]; mx = fmaxf(mx, lg[e]); }
  float den = 0.f;
  for (int e = 0; e < 8; ++e) den += __expf(lg[e] - mx);
  float inv = 1.f / den;
  for (int e = 0; e < 8; ++e) gate_o[n * 8 + e] = __expf(lg[e] - mx) * inv;
  int i0 = 0;
  for (int e = 1; e < 8; ++e) if (lg[e] > lg[i0]) i0 = e;
  int i1 = -1;
  for (int e = 0; e < 8; ++e) if (e != i0 && (i1 < 0 || lg[e] > lg[i1])) i1 = e;
  float m2 = fmaxf(lg[i0], lg[i1]);
  float e0 = __expf(lg[i0] - m2), e1 = __expf(lg[i1] - m2);
  float p0 = e0 / (e0 + e1), p1 = e1 / (e0 + e1);
  for (int e = 0; e < 8; ++e) w_all[n * 8 + e] = 0.f;
  w_all[n * 8 + i0] = p0;
  w_all[n * 8 + i1] = p1;
  routed[n]        = (i0 == 0 || i1 == 0) ? 1.f : 0.f;
  routed[2048 + n] = (i0 == 1 || i1 == 1) ? 1.f : 0.f;
  topi_o[n * 2 + 0] = (float)i0;
  topi_o[n * 2 + 1] = (float)i1;
}

// ---------------------------------------------------------------------------
// Fused residual LayerNorm (256 cols, one wave per row), optional weighted
// accumulation into the final output (h2 path).
// ---------------------------------------------------------------------------
__global__ __launch_bounds__(256) void ln_kernel(
    const float* __restrict__ a, const float* __restrict__ bsum,
    const float* __restrict__ g, const float* __restrict__ beta,
    float* __restrict__ out, float* __restrict__ acc,
    const float* __restrict__ rowscale, int rs_stride)
{
  int row  = blockIdx.x * 8 + (threadIdx.x >> 5);
  int lane = threadIdx.x & 31;
  float x[8]; float s = 0.f;
  const float* ap = a + (size_t)row * 256;
  const float* bp = bsum + (size_t)row * 256;
#pragma unroll
  for (int i = 0; i < 8; ++i) { int c = lane + i * 32; x[i] = ap[c] + bp[c]; s += x[i]; }
  s = redsum32(s);
  float mu = s * (1.f / 256.f);
  float vs = 0.f;
#pragma unroll
  for (int i = 0; i < 8; ++i) { float d = x[i] - mu; vs += d * d; }
  vs = redsum32(vs);
  float inv = rsqrtf(vs * (1.f / 256.f) + 1e-5f);
  float w = rowscale ? rowscale[(size_t)row * rs_stride] : 0.f;
#pragma unroll
  for (int i = 0; i < 8; ++i) {
    int c = lane + i * 32;
    float y = (x[i] - mu) * inv * g[c] + beta[c];
    if (out) out[(size_t)row * 256 + c] = y;
    if (acc) acc[(size_t)row * 256 + c] += w * y;
  }
}

__global__ void zero_kernel(float* __restrict__ p, int n) {
  int i = blockIdx.x * blockDim.x + threadIdx.x;
  if (i < n) p[i] = 0.f;
}

// ---------------------------------------------------------------------------
// Host-side orchestration
// ---------------------------------------------------------------------------
extern "C" void kernel_launch(void* const* d_in, const int* in_sizes, int n_in,
                              void* d_out, int out_size, void* d_ws, size_t ws_size,
                              hipStream_t stream)
{
  (void)in_sizes; (void)n_in; (void)out_size; (void)ws_size;
  const float* x    = (const float*)d_in[0];
  const float* U    = (const float*)d_in[1];
  const float* R    = (const float*)d_in[2];
  const float* S    = (const float*)d_in[3];
  const float* tp_w = (const float*)d_in[4];
  const float* tp_b = (const float*)d_in[5];
  const float* q_w  = (const float*)d_in[6];
  const float* q_b  = (const float*)d_in[7];
  const float* k_w  = (const float*)d_in[8];
  const float* k_b  = (const float*)d_in[9];
  const float* v_w  = (const float*)d_in[10];
  const float* v_b  = (const float*)d_in[11];
  const float* gwih = (const float*)d_in[12];
  const float* gwhh = (const float*)d_in[13];
  const float* gbih = (const float*)d_in[14];
  const float* gbhh = (const float*)d_in[15];
  const float* m1w  = (const float*)d_in[16];
  const float* m1b  = (const float*)d_in[17];
  const float* m2w  = (const float*)d_in[18];
  const float* m2b  = (const float*)d_in[19];
  const float* siw  = (const float*)d_in[20];
  const float* sib  = (const float*)d_in[21];
  const float* sow  = (const float*)d_in[22];
  const float* sob  = (const float*)d_in[23];
  const float* l1g  = (const float*)d_in[24];
  const float* l1b  = (const float*)d_in[25];
  const float* f1w  = (const float*)d_in[26];
  const float* f1b  = (const float*)d_in[27];
  const float* f2w  = (const float*)d_in[28];
  const float* f2b  = (const float*)d_in[29];
  const float* l2g  = (const float*)d_in[30];
  const float* l2b  = (const float*)d_in[31];
  const float* fw1  = (const float*)d_in[32];
  const float* fb1  = (const float*)d_in[33];
  const float* fw2  = (const float*)d_in[34];
  const float* fb2  = (const float*)d_in[35];

  float* outp   = (float*)d_out;            // final [2048,256]
  float* gate_o = outp + 2048 * 256;        // gating [2048,8]
  float* topi_o = gate_o + 2048 * 8;        // topi   [2048,2] (as float)

  float* ws     = (float*)d_ws;
  float* comb   = ws; ws += 2048 * 256;     // [proc | gru_out]
  float* ctxr   = ws; ws += 2048 * 32;
  float* hidden = ws; ws += 2048 * 128;
  float* logits = ws; ws += 2048 * 8;
  float* w_all  = ws; ws += 2048 * 8;
  float* routed = ws; ws += 2 * 2048;
  float* qkv    = ws; ws += 2048 * 768;
  float* ctx2   = ws; ws += 2048 * 256;
  float* attnb  = ws; ws += 2048 * 256;
  float* h1     = ws; ws += 2048 * 256;
  float* hmid   = ws; ws += 2048 * 1024;
  float* ffb    = ws; ws += 2048 * 256;

  zero_kernel<<<2048, 256, 0, stream>>>(outp, 2048 * 256);

  // ---- router: proc = relu(x @ tp_w^T + b) into comb[:, :128] ----
  gemm_wmma_kernel<<<dim3(32, 1), 128, 0, stream>>>(
      x, 256, tp_w, 256, tp_b, nullptr, 0, comb, 256, 256, GF_RELU);
  router_attn_kernel<<<256, 256, 0, stream>>>(comb, R, S, q_w, q_b, k_w, k_b, v_w, v_b, ctxr);
  gru_kernel<<<8, 384, 0, stream>>>(U, ctxr, gwih, gwhh, gbih, gbhh, comb);
  gemm_wmma_kernel<<<dim3(32, 1), 128, 0, stream>>>(
      comb, 256, m1w, 256, m1b, nullptr, 0, hidden, 128, 256, GF_RELU);
  logits_kernel<<<64, 256, 0, stream>>>(hidden, m2w, m2b, logits);
  gate_topk_kernel<<<8, 256, 0, stream>>>(logits, gate_o, topi_o, w_all, routed);

  // ---- synergy experts ----
  for (int e = 0; e < 2; ++e) {
    gemm_wmma_kernel<<<dim3(32, 6), 128, 0, stream>>>(
        x, 256, siw + (size_t)e * 768 * 256, 256, sib + e * 768,
        nullptr, 0, qkv, 768, 256, 0);
    flash_attn_kernel<<<dim3(32, 4), 128, 0, stream>>>(qkv, routed + e * 2048, ctx2, 2048);
    gemm_wmma_kernel<<<dim3(32, 2), 128, 0, stream>>>(
        ctx2, 256, sow + (size_t)e * 256 * 256, 256, sob + e * 256,
        nullptr, 0, attnb, 256, 256, 0);
    ln_kernel<<<256, 256, 0, stream>>>(x, attnb, l1g + e * 256, l1b + e * 256,
                                       h1, nullptr, nullptr, 0);
    gemm_wmma_kernel<<<dim3(32, 8), 128, 0, stream>>>(
        h1, 256, f1w + (size_t)e * 1024 * 256, 256, f1b + e * 1024,
        nullptr, 0, hmid, 1024, 256, GF_RELU);
    gemm_wmma_kernel<<<dim3(32, 2), 128, 0, stream>>>(
        hmid, 1024, f2w + (size_t)e * 256 * 1024, 1024, f2b + e * 256,
        nullptr, 0, ffb, 256, 1024, 0);
    ln_kernel<<<256, 256, 0, stream>>>(h1, ffb, l2g + e * 256, l2b + e * 256,
                                       nullptr, outp, w_all + e, 8);
  }

  // ---- dense FFN experts, weighted accumulate fused in GEMM epilogue ----
  for (int f = 0; f < 6; ++f) {
    gemm_wmma_kernel<<<dim3(32, 8), 128, 0, stream>>>(
        x, 256, fw1 + (size_t)f * 1024 * 256, 256, fb1 + f * 1024,
        nullptr, 0, hmid, 1024, 256, GF_RELU);
    gemm_wmma_kernel<<<dim3(32, 2), 128, 0, stream>>>(
        hmid, 1024, fw2 + (size_t)f * 256 * 1024, 1024, fb2 + f * 256,
        w_all + 2 + f, 8, outp, 256, 1024, GF_ACC);
  }
}
